// MultiQueryAttention_39170101739641
// MI455X (gfx1250) — matrix operand order repaired and checked by reading, not compiled, not run on hardware
//
#include <hip/hip_runtime.h>
#include <hip/hip_bf16.h>
#include <cstdint>

// MQA forward for MI455X (gfx1250, wave32, WMMA bf16 K=32).
// B=2, L=2048, D=1024, H=16, HD=64.
// d_out = [ output (B*L*D f32) | attn_probs (B*H*L*L f32) ]
// d_ws  = [ Qb bf16 (B,H,L,HD) | Kb bf16 (B,L,HD) | Vt bf16 (B,HD,L) | AO f32 (B,L,D) ]  ~25 MB

#define BB 2
#define LL 2048
#define DD 1024
#define HH 16
#define HDD 64
#define CH 64                    // key-chunk staged in LDS per block
#define NC (LL / CH)             // 32 chunks
#define SCALE 0.125f             // HD^-0.5
#define NEG_BIG (-3.402823466e38f)

typedef __bf16 bf16_t;
typedef __attribute__((ext_vector_type(16))) __bf16 v16bf;
typedef __attribute__((ext_vector_type(8)))  __bf16 v8bf;
typedef __attribute__((ext_vector_type(8)))  float  v8f;
typedef __attribute__((ext_vector_type(4)))  int    v4i;

// ---------------- async global->LDS copy (gfx1250 ASYNCcnt path) ----------------
#if __has_builtin(__builtin_amdgcn_global_load_async_to_lds_b128)
#define HAVE_ASYNC_LDS 1
#else
#define HAVE_ASYNC_LDS 0
#endif

// Builtin expects: (addrspace(1) v4i* src, addrspace(3) v4i* dst, imm offset, imm cpol)
typedef __attribute__((address_space(1))) v4i* gv4i_p;
typedef __attribute__((address_space(3))) v4i* lv4i_p;

static __device__ __forceinline__ void async_cp16(const void* g, void* l) {
#if HAVE_ASYNC_LDS
  __builtin_amdgcn_global_load_async_to_lds_b128(
      (gv4i_p)(uintptr_t)g, (lv4i_p)(uintptr_t)l, 0, 0);
#else
  *(uint4*)l = *(const uint4*)g;
#endif
}

template <int N>
static __device__ __forceinline__ void wait_async() {
#if HAVE_ASYNC_LDS
#if __has_builtin(__builtin_amdgcn_s_wait_asynccnt)
  __builtin_amdgcn_s_wait_asynccnt(N);
#else
  asm volatile("s_wait_asynccnt %0" ::"i"(N) : "memory");
#endif
#endif
}

// ---------------- small helpers ----------------
static __device__ __forceinline__ v8bf ldv8(const bf16_t* p) { return *(const v8bf*)p; }

static __device__ __forceinline__ v16bf cat8(v8bf lo, v8bf hi) {
  return __builtin_shufflevector(lo, hi, 0,1,2,3,4,5,6,7,8,9,10,11,12,13,14,15);
}

static __device__ __forceinline__ v8f bfwmma(v16bf a, v16bf b, v8f c) {
  return __builtin_amdgcn_wmma_f32_16x16x32_bf16(false, a, false, b, (short)0, c, false, false);
}

static __device__ __forceinline__ v16bf pack16(float4 a, float4 b, float4 c, float4 d) {
  v16bf r;
  r[0]=(bf16_t)a.x;  r[1]=(bf16_t)a.y;  r[2]=(bf16_t)a.z;  r[3]=(bf16_t)a.w;
  r[4]=(bf16_t)b.x;  r[5]=(bf16_t)b.y;  r[6]=(bf16_t)b.z;  r[7]=(bf16_t)b.w;
  r[8]=(bf16_t)c.x;  r[9]=(bf16_t)c.y;  r[10]=(bf16_t)c.z; r[11]=(bf16_t)c.w;
  r[12]=(bf16_t)d.x; r[13]=(bf16_t)d.y; r[14]=(bf16_t)d.z; r[15]=(bf16_t)d.w;
  return r;
}

// One 16x16 fp32-accum tile of Out = A(MxK,row-major) * W(NxK,row-major)^T, bf16 operands.
static __device__ __forceinline__ v8f gemm_tile_f32(const float* __restrict__ A, int lda,
                                                    const float* __restrict__ W, int ldw,
                                                    int row0, int col0, int K) {
  const int lane = threadIdx.x & 31;
  const int half = lane >> 4;
  const int mn   = lane & 15;
  const float* ar = A + (size_t)(row0 + mn) * lda;
  const float* wr = W + (size_t)(col0 + mn) * ldw;
  v8f acc = {};
  for (int k0 = 0; k0 < K; k0 += 32) {
    const float* ap = ar + k0 + half * 16;
    float4 a0 = *(const float4*)(ap);
    float4 a1 = *(const float4*)(ap + 4);
    float4 a2 = *(const float4*)(ap + 8);
    float4 a3 = *(const float4*)(ap + 12);
    const float* wp = wr + k0 + half * 16;
    float4 b0 = *(const float4*)(wp);
    float4 b1 = *(const float4*)(wp + 4);
    float4 b2 = *(const float4*)(wp + 8);
    float4 b3 = *(const float4*)(wp + 12);
    acc = bfwmma(pack16(a0, a1, a2, a3), pack16(b0, b1, b2, b3), acc);
  }
  return acc;
}

// ---------------- Kernel A: fused Q/K/V projections (fp32 in, bf16 out) ----------------
__global__ void __launch_bounds__(256)
qkv_proj_kernel(const float* __restrict__ query, const float* __restrict__ key,
                const float* __restrict__ value, const float* __restrict__ Wq,
                const float* __restrict__ Wk, const float* __restrict__ Wv,
                bf16_t* __restrict__ Qb, bf16_t* __restrict__ Kb, bf16_t* __restrict__ Vt) {
  const int wid  = (int)((blockIdx.x * blockDim.x + threadIdx.x) >> 5);
  const int lane = threadIdx.x & 31;
  const int half = lane >> 4;
  const int mn   = lane & 15;
  const int NQ  = BB * (LL / 16) * (DD / 16);   // 16384
  const int NKV = BB * (LL / 16) * (HDD / 16);  // 1024

  if (wid < NQ) {
    const int per_b = (LL / 16) * (DD / 16);
    const int b = wid / per_b, r = wid % per_b;
    const int mt = r / (DD / 16), nt = r % (DD / 16);
    v8f c = gemm_tile_f32(query + (size_t)b * LL * DD, DD, Wq, DD, mt * 16, nt * 16, DD);
    const int n = nt * 16 + mn, h = n >> 6, hd = n & 63;
#pragma unroll
    for (int v = 0; v < 8; ++v) {
      const int row = mt * 16 + v + half * 8;
      Qb[(((size_t)b * HH + h) * LL + row) * HDD + hd] = (bf16_t)c[v];
    }
  } else if (wid < NQ + NKV) {
    const int w = wid - NQ;
    const int per_b = (LL / 16) * (HDD / 16);
    const int b = w / per_b, r = w % per_b;
    const int mt = r / (HDD / 16), nt = r % (HDD / 16);
    v8f c = gemm_tile_f32(key + (size_t)b * LL * DD, DD, Wk, DD, mt * 16, nt * 16, DD);
    const int n = nt * 16 + mn;
#pragma unroll
    for (int v = 0; v < 8; ++v) {
      const int row = mt * 16 + v + half * 8;
      Kb[((size_t)b * LL + row) * HDD + n] = (bf16_t)c[v];
    }
  } else if (wid < NQ + 2 * NKV) {
    const int w = wid - NQ - NKV;
    const int per_b = (LL / 16) * (HDD / 16);
    const int b = w / per_b, r = w % per_b;
    const int mt = r / (HDD / 16), nt = r % (HDD / 16);
    v8f c = gemm_tile_f32(value + (size_t)b * LL * DD, DD, Wv, DD, mt * 16, nt * 16, DD);
    const int n = nt * 16 + mn;
#pragma unroll
    for (int v = 0; v < 8; ++v) {
      const int row = mt * 16 + v + half * 8;
      Vt[((size_t)b * HDD + n) * LL + row] = (bf16_t)c[v];  // (B,HD,L)
    }
  }
}

// ---------------- Kernel B: attention ----------------
// Block = 8 waves sharing batch b; each wave owns one (h, 16-row q-tile).
// K (and V in emit sweep) chunks are async-staged into LDS once per block and
// consumed by all 8 waves. Softmax stats are kept per-lane during the sweep
// (no shuffles) and merged across lanes once.
__global__ void __launch_bounds__(256)
attn_kernel(const bf16_t* __restrict__ Qb, const bf16_t* __restrict__ Kb,
            const bf16_t* __restrict__ Vt, const uint8_t* __restrict__ mask,
            float* __restrict__ probs, float* __restrict__ AO) {
  // Padded pitches (72 bf16 = 144 B = 36 dwords) spread the 64 LDS banks conflict-free.
  __shared__ __align__(16) bf16_t Kch[2][CH][72];   // 18 KB
  __shared__ __align__(16) bf16_t Vch[2][HDD][72];  // 18 KB
  __shared__ __align__(16) bf16_t pbuf[8][16][40];  // 10 KB P bounce (C-layout -> A-layout)
  __shared__ uint8_t mlds[LL];                      // 2 KB mask row

  const int tid  = threadIdx.x;
  const int wib  = tid >> 5;
  const int lane = tid & 31;
  const int half = lane >> 4;
  const int mn   = lane & 15;

  const int b   = blockIdx.x >> 8;                 // 256 blocks per batch
  const int job = ((blockIdx.x & 255) << 3) | wib; // 0..2047
  const int h   = job >> 7;
  const int q0  = (job & 127) * 16;

  const bf16_t*  kbase = Kb + (size_t)b * LL * HDD;
  const bf16_t*  vbase = Vt + (size_t)b * HDD * LL;
  const uint8_t* mrow  = mask + (size_t)b * LL;

  // stage the mask row (2 KB) once
  ((uint2*)mlds)[tid] = ((const uint2*)mrow)[tid];

  // Q A-fragments for hd-chunks [0,32) and [32,64); live in regs for the whole kernel.
  const bf16_t* qrow = Qb + (((size_t)b * HH + h) * LL + q0 + mn) * HDD;
  v16bf aq0 = cat8(ldv8(qrow + half * 16),      ldv8(qrow + 8 + half * 16));
  v16bf aq1 = cat8(ldv8(qrow + 32 + half * 16), ldv8(qrow + 40 + half * 16));

  // ---- chunk copy helpers (per-thread slices; 2 b128 issues each) ----
  auto load_k_chunk = [&](int kc0, int buf) {
#pragma unroll
    for (int i = 0; i < 2; ++i) {
      const int idx = i * 256 + tid;     // 0..511
      const int row = idx >> 3;          // 0..63 keys
      const int seg = idx & 7;           // 8 x 16B per 128B row
      async_cp16(kbase + (size_t)(kc0 + row) * HDD + seg * 8, &Kch[buf][row][seg * 8]);
    }
  };
  auto load_v_chunk = [&](int kc0, int buf) {
#pragma unroll
    for (int i = 0; i < 2; ++i) {
      const int idx = i * 256 + tid;
      const int hd  = idx >> 3;          // 0..63
      const int seg = idx & 7;
      async_cp16(vbase + (size_t)hd * LL + kc0 + seg * 8, &Vch[buf][hd][seg * 8]);
    }
  };

  float m[8], ssum[8];
#pragma unroll
  for (int v = 0; v < 8; ++v) { m[v] = NEG_BIG; ssum[v] = 0.f; }

  // ================= Sweep 1: per-lane online (max,sum) stats =================
  load_k_chunk(0, 0);
  for (int c = 0; c < NC; ++c) {
    if (c + 1 < NC) { load_k_chunk((c + 1) * CH, (c + 1) & 1); wait_async<2>(); }
    else            { wait_async<0>(); }
    __syncthreads();
    const bf16_t (*Kc)[72] = Kch[c & 1];
    const int kc0 = c * CH;
#pragma unroll
    for (int kt = 0; kt < CH / 16; ++kt) {
      const int lkey = kt * 16 + mn;
      v16bf kb0 = cat8(ldv8(&Kc[lkey][half * 16]),      ldv8(&Kc[lkey][half * 16 + 8]));
      v16bf kb1 = cat8(ldv8(&Kc[lkey][32 + half * 16]), ldv8(&Kc[lkey][40 + half * 16]));
      v8f s = {};
      s = bfwmma(aq0, kb0, s);
      s = bfwmma(aq1, kb1, s);
      const bool mb = mlds[kc0 + lkey] != 0;
#pragma unroll
      for (int v = 0; v < 8; ++v) {
        const float sv = mb ? s[v] * SCALE : NEG_BIG;
        const float nm = fmaxf(m[v], sv);
        ssum[v] = ssum[v] * __expf(m[v] - nm) + __expf(sv - nm);
        m[v] = nm;
      }
    }
    __syncthreads();
  }

  // ---- merge (m, ssum) pairs across the 16 lanes of each half (4 steps, once) ----
#pragma unroll
  for (int v = 0; v < 8; ++v) {
#pragma unroll
    for (int msk = 1; msk <= 8; msk <<= 1) {
      const float om = __shfl_xor(m[v], msk, 32);
      const float os = __shfl_xor(ssum[v], msk, 32);
      const float nm = fmaxf(m[v], om);
      ssum[v] = ssum[v] * __expf(m[v] - nm) + os * __expf(om - nm);
      m[v] = nm;
    }
  }
  float inv[8];
#pragma unroll
  for (int v = 0; v < 8; ++v) inv[v] = 1.0f / ssum[v];

  // ================= Sweep 2: write normalized probs + fused P.V =================
  bf16_t (*pw)[40] = pbuf[wib];
  v8f o0 = {}, o1 = {}, o2 = {}, o3 = {};
  const size_t prow0 = (((size_t)b * HH + h) * LL + q0) * LL;

  load_k_chunk(0, 0);
  load_v_chunk(0, 0);
  for (int c = 0; c < NC; ++c) {
    if (c + 1 < NC) {
      load_k_chunk((c + 1) * CH, (c + 1) & 1);
      load_v_chunk((c + 1) * CH, (c + 1) & 1);
      wait_async<4>();
    } else {
      wait_async<0>();
    }
    __syncthreads();
    const bf16_t (*Kc)[72] = Kch[c & 1];
    const bf16_t (*Vc)[72] = Vch[c & 1];
    const int kc0 = c * CH;
#pragma unroll
    for (int sub = 0; sub < CH / 32; ++sub) {
#pragma unroll
      for (int t = 0; t < 2; ++t) {
        const int lkey = sub * 32 + t * 16 + mn;
        v16bf kb0 = cat8(ldv8(&Kc[lkey][half * 16]),      ldv8(&Kc[lkey][half * 16 + 8]));
        v16bf kb1 = cat8(ldv8(&Kc[lkey][32 + half * 16]), ldv8(&Kc[lkey][40 + half * 16]));
        v8f s = {};
        s = bfwmma(aq0, kb0, s);
        s = bfwmma(aq1, kb1, s);
        const bool mb = mlds[kc0 + lkey] != 0;
#pragma unroll
        for (int v = 0; v < 8; ++v) {
          const float sv = mb ? s[v] * SCALE : NEG_BIG;
          const float p  = __expf(sv - m[v]) * inv[v];
          const int row  = v + 8 * half;
          probs[prow0 + (size_t)row * LL + kc0 + lkey] = p;  // normalized, written once
          pw[row][t * 16 + mn] = (bf16_t)p;                  // C-layout -> LDS
        }
      }
      asm volatile("s_wait_dscnt 0" ::: "memory");  // order the per-wave LDS bounce
      v16bf pA = cat8(ldv8(&pw[mn][half * 16]), ldv8(&pw[mn][half * 16 + 8]));
      const int ko = sub * 32 + half * 16;
      v16bf vb0 = cat8(ldv8(&Vc[ 0 + mn][ko]), ldv8(&Vc[ 0 + mn][ko + 8]));
      v16bf vb1 = cat8(ldv8(&Vc[16 + mn][ko]), ldv8(&Vc[16 + mn][ko + 8]));
      v16bf vb2 = cat8(ldv8(&Vc[32 + mn][ko]), ldv8(&Vc[32 + mn][ko + 8]));
      v16bf vb3 = cat8(ldv8(&Vc[48 + mn][ko]), ldv8(&Vc[48 + mn][ko + 8]));
      o0 = bfwmma(pA, vb0, o0);
      o1 = bfwmma(pA, vb1, o1);
      o2 = bfwmma(pA, vb2, o2);
      o3 = bfwmma(pA, vb3, o3);
    }
    __syncthreads();
  }

  // attn_out (B,L,D) fp32; d = h*64 + j*16 + n
#pragma unroll
  for (int v = 0; v < 8; ++v) {
    const size_t rowb = ((size_t)b * LL + q0 + v + 8 * half) * DD + (size_t)h * HDD + mn;
    AO[rowb + 0]  = o0[v];
    AO[rowb + 16] = o1[v];
    AO[rowb + 32] = o2[v];
    AO[rowb + 48] = o3[v];
  }
}

// ---------------- Kernel C: output projection  out = AO @ Wo^T + bo ----------------
__global__ void __launch_bounds__(256)
out_proj_kernel(const float* __restrict__ AO, const float* __restrict__ Wo,
                const float* __restrict__ bo, float* __restrict__ out) {
  const int wid  = (int)((blockIdx.x * blockDim.x + threadIdx.x) >> 5);
  const int lane = threadIdx.x & 31;
  const int half = lane >> 4;
  const int mn   = lane & 15;
  const int mt = wid / (DD / 16);
  const int nt = wid % (DD / 16);
  v8f c = gemm_tile_f32(AO, DD, Wo, DD, mt * 16, nt * 16, DD);
  const int n = nt * 16 + mn;
  const float bias = bo[n];
#pragma unroll
  for (int v = 0; v < 8; ++v) {
    const size_t row = (size_t)mt * 16 + v + 8 * half;
    out[row * DD + n] = c[v] + bias;
  }
}

extern "C" void kernel_launch(void* const* d_in, const int* in_sizes, int n_in,
                              void* d_out, int out_size, void* d_ws, size_t ws_size,
                              hipStream_t stream) {
  (void)in_sizes; (void)n_in; (void)out_size; (void)ws_size;
  const float*   query = (const float*)d_in[0];
  const float*   key   = (const float*)d_in[1];
  const float*   value = (const float*)d_in[2];
  const uint8_t* mask  = (const uint8_t*)d_in[3];
  const float*   Wq    = (const float*)d_in[4];
  const float*   Wk    = (const float*)d_in[5];
  const float*   Wv    = (const float*)d_in[6];
  const float*   Wo    = (const float*)d_in[7];
  const float*   bo    = (const float*)d_in[8];

  float* out   = (float*)d_out;
  float* probs = out + (size_t)BB * LL * DD;

  bf16_t* Qb = (bf16_t*)d_ws;                    //  8 MB: (B,H,L,HD)
  bf16_t* Kb = Qb + (size_t)BB * HH * LL * HDD;  // 0.5 MB: (B,L,HD)
  bf16_t* Vt = Kb + (size_t)BB * LL * HDD;       // 0.5 MB: (B,HD,L)
  float*  AO = (float*)(Vt + (size_t)BB * LL * HDD);  // 16 MB: (B,L,D)

  qkv_proj_kernel<<<2304, 256, 0, stream>>>(query, key, value, Wq, Wk, Wv, Qb, Kb, Vt);
  attn_kernel<<<512, 256, 0, stream>>>(Qb, Kb, Vt, mask, probs, AO);
  out_proj_kernel<<<2048, 256, 0, stream>>>(AO, Wo, bo, out);
}
